// STMambaBlock_24395414241540
// MI455X (gfx1250) — compile-verified, matrix operand order checked by reading, not run on hardware
//
#include <hip/hip_runtime.h>
#include <hip/hip_bf16.h>
#include <math.h>

#define DIM     192
#define DSTATE  16
#define DTRANK  12
#define DINNER  384
#define XZW     768   // 2*DINNER
#define MLPHID  768
#define BB      2
#define TT      8
#define NSP     576   // H*W
#define NTOK    (BB*TT*NSP)  // 9216
#define KMAX    768

typedef __attribute__((ext_vector_type(16))) _Float16 v16h;
typedef __attribute__((ext_vector_type(8)))  _Float16 v8h_t;
typedef __attribute__((ext_vector_type(8)))  float    v8f;
typedef __attribute__((ext_vector_type(4)))  unsigned int u32x4;
typedef __attribute__((ext_vector_type(8)))  int          i32x8;
typedef __attribute__((ext_vector_type(4)))  int          i32x4;

__device__ __forceinline__ float wred32(float v) {
#pragma unroll
  for (int o = 16; o > 0; o >>= 1) v += __shfl_xor(v, o, 32);
  return v;
}

__device__ __forceinline__ float silu_f(float x) { return x / (1.f + __expf(-x)); }

// ---------------------------------------------------------------- f32 -> f16
__global__ __launch_bounds__(256) void cvt_f16_kernel(const float* __restrict__ s,
                                                      _Float16* __restrict__ d, int n) {
  int i = blockIdx.x * 256 + threadIdx.x;
  if (i < n) d[i] = (_Float16)s[i];
}

// ---------------------------------------------------- norm1: transpose + LN
// x_in [BT, C, N] -> shortcut f32 [tok, C], xn f16 [tok, C]
__global__ __launch_bounds__(256) void norm1_kernel(const float* __restrict__ xin,
                                                    const float* __restrict__ w,
                                                    const float* __restrict__ b,
                                                    float* __restrict__ sc,
                                                    _Float16* __restrict__ xn) {
  int r = blockIdx.x * 8 + (threadIdx.x >> 5);
  int lane = threadIdx.x & 31;
  int bt = r / NSP, n = r - bt * NSP;
  const float* base = xin + (size_t)bt * DIM * NSP + n;
  float v[6];
#pragma unroll
  for (int i = 0; i < 6; i++) { int c = lane + 32 * i; v[i] = base[(size_t)c * NSP]; }
  float s = 0.f;
#pragma unroll
  for (int i = 0; i < 6; i++) s += v[i];
  float m = wred32(s) * (1.f / DIM);
  float q = 0.f;
#pragma unroll
  for (int i = 0; i < 6; i++) { float d0 = v[i] - m; q += d0 * d0; }
  float rs = rsqrtf(wred32(q) * (1.f / DIM) + 1e-5f);
#pragma unroll
  for (int i = 0; i < 6; i++) {
    int c = lane + 32 * i;
    sc[(size_t)r * DIM + c] = v[i];
    xn[(size_t)r * DIM + c] = (_Float16)((v[i] - m) * rs * w[c] + b[c]);
  }
}

// ------------------------------------------------------------ norm2 (rowmaj)
__global__ __launch_bounds__(256) void norm2_kernel(const float* __restrict__ x2,
                                                    const float* __restrict__ w,
                                                    const float* __restrict__ b,
                                                    _Float16* __restrict__ hn) {
  int r = blockIdx.x * 8 + (threadIdx.x >> 5);
  int lane = threadIdx.x & 31;
  const float* row = x2 + (size_t)r * DIM;
  float v[6];
#pragma unroll
  for (int i = 0; i < 6; i++) v[i] = row[lane + 32 * i];
  float s = 0.f;
#pragma unroll
  for (int i = 0; i < 6; i++) s += v[i];
  float m = wred32(s) * (1.f / DIM);
  float q = 0.f;
#pragma unroll
  for (int i = 0; i < 6; i++) { float d0 = v[i] - m; q += d0 * d0; }
  float rs = rsqrtf(wred32(q) * (1.f / DIM) + 1e-5f);
#pragma unroll
  for (int i = 0; i < 6; i++) {
    int c = lane + 32 * i;
    hn[(size_t)r * DIM + c] = (_Float16)((v[i] - m) * rs * w[c] + b[c]);
  }
}

// --------------------------------------------- TDM: stage 16xK f16 panel in LDS
// D# per CDNA5 ISA 8.3/8.4: group0 = count/lds_addr/global_addr/type,
// group1 = data_size=2B, tensor_dim0=K, tensor_dim1=16, tile 16xK, stride K.
// This toolchain exposes the 6-arg builtin: (g0, g1, g2, g3, g_extra, cpol).
__device__ __forceinline__ void tdm_load_b_panel(const _Float16* gsrc,
                                                 unsigned lds_off, int K) {
  unsigned long long ga = (unsigned long long)(uintptr_t)gsrc;
  u32x4 g0;
  g0[0] = 1u;                                   // count=1 (valid user descriptor)
  g0[1] = lds_off;                              // lds_addr (bytes)
  g0[2] = (unsigned)(ga & 0xffffffffu);         // global_addr[31:0]
  g0[3] = (unsigned)((ga >> 32) & 0x01ffffffu)  // global_addr[56:32]
          | (2u << 30);                         // type = 2 ("image")
  unsigned uK = (unsigned)K;
  i32x8 g1;
  g1[0] = (int)(1u << 16);                      // data_size=1 -> 2 bytes
  g1[1] = (int)((uK & 0xffffu) << 16);          // tensor_dim0[15:0] @bit48
  g1[2] = (int)((uK >> 16) | (16u << 16));      // tensor_dim0[31:16] | tensor_dim1=16
  g1[3] = (int)(uK << 16);                      // tensor_dim1 hi=0 | tile_dim0=K
  g1[4] = 16;                                   // tile_dim1=16 | tile_dim2=0
  g1[5] = (int)uK;                              // tensor_dim0_stride[31:0] = K
  g1[6] = 0;                                    // stride[47:32] | dim1_stride lo
  g1[7] = 0;
  i32x4 z4 = {0, 0, 0, 0};                      // groups 2/3 unused (2-D tensor)
  i32x8 z8 = {0, 0, 0, 0, 0, 0, 0, 0};
  __builtin_amdgcn_tensor_load_to_lds(g0, g1, z4, z4, z8, 0);
}

// ------------------------------------------------------- WMMA tile mainloop
// A [M,K] row-major f16 (global); B panel 16xK f16 in LDS (row n contiguous).
__device__ __forceinline__ v8f wmma_tile(const _Float16* __restrict__ A, int lda,
                                         const _Float16* bl,  // LDS, + ln*K + hi*8
                                         int K, int tm, int lane) {
  const int hi = lane >> 4;
  const int ln = lane & 15;
  const _Float16* ap = A + (size_t)(tm * 16 + ln) * lda + hi * 8;
  v8f acc = {};
  for (int kb = 0; kb < K; kb += 32) {
    union { v16h v; v8h_t h[2]; } a, b;
    a.h[0] = *(const v8h_t*)(ap + kb);
    a.h[1] = *(const v8h_t*)(ap + kb + 16);
    b.h[0] = *(const v8h_t*)(bl + kb);          // ds_load_b128
    b.h[1] = *(const v8h_t*)(bl + kb + 16);
    if (kb + 32 < K) {                          // CDNA5 global_prefetch_b8
      __builtin_prefetch(ap + kb + 32, 0, 0);
    }
    acc = __builtin_amdgcn_wmma_f32_16x16x32_f16(false, a.v, false, b.v,
                                                 (short)0, acc, false, false);
  }
  return acc;
}

// Grid: blockIdx.x = mgroup * tilesN + tn ; wave w computes tile (mgroup*8+w, tn).
// All 8 waves share one 16xK weight panel, TDM-staged into LDS once per block.
// EPI: 0 = f16 store; 2 = +bias, exact gelu, f16 store;
//      3 = +bias+resid, f32 store; 4 = +bias+resid, transposed f32 store (final)
template <int EPI>
__global__ __launch_bounds__(256) void wmma_gemm_kernel(
    const _Float16* __restrict__ A, int lda, const _Float16* __restrict__ B, int ldb,
    int M, int N, int K, void* __restrict__ Cout, int ldc, int ccol,
    const float* __restrict__ bias, const float* __restrict__ resid, int ldr) {
  __shared__ __align__(16) _Float16 bsm[16 * KMAX];
  int tilesN = N >> 4;
  int tn = blockIdx.x % tilesN;
  int mg = blockIdx.x / tilesN;
  int wave = threadIdx.x >> 5;
  int lane = threadIdx.x & 31;
  int tm = mg * 8 + wave;

  if (threadIdx.x < 32) {                       // one wave drives the TDM DMA
    tdm_load_b_panel(B + (size_t)tn * 16 * ldb,
                     (unsigned)(uintptr_t)&bsm[0], K);
    __builtin_amdgcn_s_wait_tensorcnt(0);
  }
  __syncthreads();

  int hi = lane >> 4, ln = lane & 15;
  v8f acc = wmma_tile(A, lda, bsm + ln * K + hi * 8, K, tm, lane);

  int col = tn * 16 + ln;
#pragma unroll
  for (int v = 0; v < 8; v++) {
    int row = tm * 16 + v + hi * 8;
    float x = acc[v];
    if (EPI == 0) {
      ((_Float16*)Cout)[(size_t)row * ldc + ccol + col] = (_Float16)x;
    } else if (EPI == 2) {
      x += bias[col];
      x = 0.5f * x * (1.f + erff(x * 0.70710678118f));
      ((_Float16*)Cout)[(size_t)row * ldc + col] = (_Float16)x;
    } else if (EPI == 3) {
      x += bias[col] + resid[(size_t)row * ldr + col];
      ((float*)Cout)[(size_t)row * ldc + col] = x;
    } else {
      x += bias[col] + resid[(size_t)row * ldr + col];
      int bt = row / NSP, nn = row - bt * NSP;
      ((float*)Cout)[((size_t)bt * DIM + col) * NSP + nn] = x;  // [B,T,C,H,W]
    }
  }
}

// -------------------------------------------- causal depthwise conv1d + silu
__global__ __launch_bounds__(256) void conv_silu_kernel(
    const _Float16* __restrict__ xz, const float* __restrict__ cw,
    const float* __restrict__ cb, float* __restrict__ u,
    int pos_div, int pos_mod, int tok_stride) {
  int idx = blockIdx.x * 256 + threadIdx.x;
  if (idx >= NTOK * DINNER) return;
  int r = idx / DINNER, d = idx - r * DINNER;
  int pos = (r / pos_div) % pos_mod;
  float acc = cb[d];
#pragma unroll
  for (int j = 0; j < 4; j++) {
    int p = pos - 3 + j;
    if (p >= 0)
      acc += cw[d * 4 + j] * (float)xz[(size_t)(r + (p - pos) * tok_stride) * XZW + d];
  }
  u[(size_t)r * DINNER + d] = silu_f(acc);
}

// ------------------------------- x_proj (44 outs) + dt_proj + softplus, per token
__global__ __launch_bounds__(256) void xdt_prep_kernel(
    const float* __restrict__ u, const float* __restrict__ xpw,   // [44,384]
    const float* __restrict__ dtw, const float* __restrict__ dtb, // [384,12],[384]
    float* __restrict__ delta, float* __restrict__ Bm, float* __restrict__ Cm) {
  int r = blockIdx.x * 8 + (threadIdx.x >> 5);
  int lane = threadIdx.x & 31;
  const float* ur = u + (size_t)r * DINNER;
  float dtv[DTRANK];
#pragma unroll
  for (int o = 0; o < DTRANK + 2 * DSTATE; o++) {
    const float* wrow = xpw + o * DINNER;
    float s = 0.f;
#pragma unroll
    for (int k = 0; k < 12; k++) s += ur[lane + 32 * k] * wrow[lane + 32 * k];
    s = wred32(s);
    if (o < DTRANK) { dtv[o] = s; }
    else if (o < DTRANK + DSTATE) { if (lane == 0) Bm[(size_t)r * DSTATE + (o - DTRANK)] = s; }
    else { if (lane == 0) Cm[(size_t)r * DSTATE + (o - DTRANK - DSTATE)] = s; }
  }
#pragma unroll
  for (int i = 0; i < 12; i++) {
    int d = lane + 32 * i;
    float a = dtb[d];
#pragma unroll
    for (int j = 0; j < DTRANK; j++) a += dtv[j] * dtw[d * DTRANK + j];
    delta[(size_t)r * DINNER + d] = (a > 20.f) ? a : log1pf(__expf(a));  // softplus
  }
}

// ----------------------------------------------- selective scan (+silu gate)
// one thread per (sequence, channel d); 16 states in registers
__global__ __launch_bounds__(384) void scan_kernel(
    const float* __restrict__ u, const float* __restrict__ delta,
    const float* __restrict__ Bm, const float* __restrict__ Cm,
    const _Float16* __restrict__ xz, const float* __restrict__ A_log,
    const float* __restrict__ Dv, _Float16* __restrict__ ys,
    int L, int sdiv, int mulA, int mulB, int stride) {
  int d = threadIdx.x;
  int seq = blockIdx.x;
  int base = (seq / sdiv) * mulA + (seq % sdiv) * mulB;
  float Ad[DSTATE], h[DSTATE];
#pragma unroll
  for (int n = 0; n < DSTATE; n++) { Ad[n] = -__expf(A_log[d * DSTATE + n]); h[n] = 0.f; }
  float Dd = Dv[d];
  for (int p = 0; p < L; p++) {
    size_t r = (size_t)base + (size_t)p * stride;
    float dl = delta[r * DINNER + d];
    float uu = u[r * DINNER + d];
    float du = dl * uu;
    const float* bp = Bm + r * DSTATE;
    const float* cp = Cm + r * DSTATE;
    float y = 0.f;
#pragma unroll
    for (int n = 0; n < DSTATE; n++) {
      float dA = __expf(dl * Ad[n]);
      h[n] = fmaf(dA, h[n], du * bp[n]);
      y = fmaf(h[n], cp[n], y);
    }
    y = fmaf(uu, Dd, y);
    float z = (float)xz[r * XZW + DINNER + d];
    y *= silu_f(z);
    ys[r * DINNER + d] = (_Float16)y;
  }
}

// ============================================================================
extern "C" void kernel_launch(void* const* d_in, const int* in_sizes, int n_in,
                              void* d_out, int out_size, void* d_ws, size_t ws_size,
                              hipStream_t stream) {
  const float* x_in = (const float*)d_in[0];
  const float* n1w = (const float*)d_in[1];
  const float* n1b = (const float*)d_in[2];
  const float* s_inw = (const float*)d_in[3];
  const float* s_cw  = (const float*)d_in[4];
  const float* s_cb  = (const float*)d_in[5];
  const float* s_xp  = (const float*)d_in[6];
  const float* s_dtw = (const float*)d_in[7];
  const float* s_dtb = (const float*)d_in[8];
  const float* s_Al  = (const float*)d_in[9];
  const float* s_D   = (const float*)d_in[10];
  const float* s_ow  = (const float*)d_in[11];
  const float* t_inw = (const float*)d_in[12];
  const float* t_cw  = (const float*)d_in[13];
  const float* t_cb  = (const float*)d_in[14];
  const float* t_xp  = (const float*)d_in[15];
  const float* t_dtw = (const float*)d_in[16];
  const float* t_dtb = (const float*)d_in[17];
  const float* t_Al  = (const float*)d_in[18];
  const float* t_D   = (const float*)d_in[19];
  const float* t_ow  = (const float*)d_in[20];
  const float* fw    = (const float*)d_in[21];
  const float* fb    = (const float*)d_in[22];
  const float* n2w   = (const float*)d_in[23];
  const float* n2b   = (const float*)d_in[24];
  const float* w1    = (const float*)d_in[25];
  const float* b1    = (const float*)d_in[26];
  const float* w2    = (const float*)d_in[27];
  const float* b2    = (const float*)d_in[28];

  char* ws = (char*)d_ws;
  size_t off = 0;
  auto alloc = [&](size_t bytes) -> void* {
    void* p = ws + off;
    off = (off + bytes + 255) & ~(size_t)255;
    return p;
  };
  _Float16* w16_sin  = (_Float16*)alloc((size_t)XZW * DIM * 2);
  _Float16* w16_tin  = (_Float16*)alloc((size_t)XZW * DIM * 2);
  _Float16* w16_sout = (_Float16*)alloc((size_t)DIM * DINNER * 2);
  _Float16* w16_tout = (_Float16*)alloc((size_t)DIM * DINNER * 2);
  _Float16* w16_fus  = (_Float16*)alloc((size_t)DIM * DINNER * 2);
  _Float16* w16_w1   = (_Float16*)alloc((size_t)MLPHID * DIM * 2);
  _Float16* w16_w2   = (_Float16*)alloc((size_t)DIM * MLPHID * 2);
  float*    sc    = (float*)alloc((size_t)NTOK * DIM * 4);      // shortcut
  _Float16* xn    = (_Float16*)alloc((size_t)NTOK * DIM * 2);   // also hn2 later
  _Float16* xz    = (_Float16*)alloc((size_t)NTOK * XZW * 2);   // also mlp h1 later
  float*    u     = (float*)alloc((size_t)NTOK * DINNER * 4);
  float*    delta = (float*)alloc((size_t)NTOK * DINNER * 4);
  float*    Bm    = (float*)alloc((size_t)NTOK * DSTATE * 4);
  float*    Cm    = (float*)alloc((size_t)NTOK * DSTATE * 4);
  _Float16* ysb   = (_Float16*)alloc((size_t)NTOK * DINNER * 2);
  _Float16* fused = (_Float16*)alloc((size_t)NTOK * DINNER * 2);
  float*    x2    = (float*)alloc((size_t)NTOK * DIM * 4);
  (void)ws_size; (void)n_in; (void)in_sizes; (void)out_size;

  auto cvt = [&](const float* s, _Float16* d, int n) {
    cvt_f16_kernel<<<(n + 255) / 256, 256, 0, stream>>>(s, d, n);
  };
  cvt(s_inw, w16_sin, XZW * DIM);
  cvt(t_inw, w16_tin, XZW * DIM);
  cvt(s_ow, w16_sout, DIM * DINNER);
  cvt(t_ow, w16_tout, DIM * DINNER);
  cvt(fw, w16_fus, DIM * DINNER);
  cvt(w1, w16_w1, MLPHID * DIM);
  cvt(w2, w16_w2, DIM * MLPHID);

  const int nblkTok = NTOK / 8;  // wave-per-token kernels
  norm1_kernel<<<nblkTok, 256, 0, stream>>>(x_in, n1w, n1b, sc, xn);

  // blocks = tilesN * (tilesM/8); tn fixed per block (shared TDM weight panel)
  auto gemm_blocks = [](int M, int N) { return (N >> 4) * ((M >> 4) / 8); };

  // ---------------- spatial mamba ----------------
  wmma_gemm_kernel<0><<<gemm_blocks(NTOK, XZW), 256, 0, stream>>>(
      xn, DIM, w16_sin, DIM, NTOK, XZW, DIM, xz, XZW, 0, nullptr, nullptr, 0);
  conv_silu_kernel<<<(NTOK * DINNER + 255) / 256, 256, 0, stream>>>(
      xz, s_cw, s_cb, u, 1, NSP, 1);
  xdt_prep_kernel<<<nblkTok, 256, 0, stream>>>(u, s_xp, s_dtw, s_dtb, delta, Bm, Cm);
  scan_kernel<<<BB * TT, DINNER, 0, stream>>>(u, delta, Bm, Cm, xz, s_Al, s_D, ysb,
                                              NSP, 1, NSP, 0, 1);
  wmma_gemm_kernel<0><<<gemm_blocks(NTOK, DIM), 256, 0, stream>>>(
      ysb, DINNER, w16_sout, DINNER, NTOK, DIM, DINNER, fused, DINNER, 0,
      nullptr, nullptr, 0);

  // ---------------- temporal mamba (reuses xz/u/delta/Bm/Cm/ysb) ----------------
  wmma_gemm_kernel<0><<<gemm_blocks(NTOK, XZW), 256, 0, stream>>>(
      xn, DIM, w16_tin, DIM, NTOK, XZW, DIM, xz, XZW, 0, nullptr, nullptr, 0);
  conv_silu_kernel<<<(NTOK * DINNER + 255) / 256, 256, 0, stream>>>(
      xz, t_cw, t_cb, u, NSP, TT, NSP);
  xdt_prep_kernel<<<nblkTok, 256, 0, stream>>>(u, t_xp, t_dtw, t_dtb, delta, Bm, Cm);
  scan_kernel<<<BB * NSP, DINNER, 0, stream>>>(u, delta, Bm, Cm, xz, t_Al, t_D, ysb,
                                               TT, NSP, TT * NSP, 1, NSP);
  wmma_gemm_kernel<0><<<gemm_blocks(NTOK, DIM), 256, 0, stream>>>(
      ysb, DINNER, w16_tout, DINNER, NTOK, DIM, DINNER, fused, DINNER, DIM,
      nullptr, nullptr, 0);

  // ---------------- fusion + residual ----------------
  wmma_gemm_kernel<3><<<gemm_blocks(NTOK, DIM), 256, 0, stream>>>(
      fused, DINNER, w16_fus, DINNER, NTOK, DIM, DINNER, x2, DIM, 0, fb, sc, DIM);

  // ---------------- MLP ----------------
  _Float16* hn2 = xn;   // reuse (xn fully consumed)
  _Float16* h1  = xz;   // reuse (xz fully consumed)
  norm2_kernel<<<nblkTok, 256, 0, stream>>>(x2, n2w, n2b, hn2);
  wmma_gemm_kernel<2><<<gemm_blocks(NTOK, MLPHID), 256, 0, stream>>>(
      hn2, DIM, w16_w1, DIM, NTOK, MLPHID, DIM, h1, MLPHID, 0, b1, nullptr, 0);
  wmma_gemm_kernel<4><<<gemm_blocks(NTOK, DIM), 256, 0, stream>>>(
      h1, MLPHID, w16_w2, MLPHID, NTOK, DIM, MLPHID, d_out, DIM, 0, b2, x2, DIM);
}
